// LearnedTransitionAttention_43224550867972
// MI455X (gfx1250) — matrix operand order repaired and checked by reading, not compiled, not run on hardware
//
#include <hip/hip_runtime.h>

static constexpr int CB = 2, CT = 2048, CD = 1024, CH = 16, CDH = 64;
static constexpr size_t NBT = (size_t)CB * CT;          // 4096 rows
static constexpr size_t NXD = NBT * CD;                 // 4,194,304

typedef unsigned short u16;
typedef __attribute__((ext_vector_type(16))) __bf16 v16bf;
typedef __attribute__((ext_vector_type(8)))  float  v8f;

// ---------------- helpers ----------------
__device__ __forceinline__ u16 f2bf(float f) {
  unsigned x = __float_as_uint(f);
  unsigned r = (x + 0x7FFFu + ((x >> 16) & 1u)) >> 16;   // RNE
  return (u16)r;
}

__device__ __forceinline__ v8f v8f_zero() {
  union { v8f v; float f[8]; } z;
#pragma unroll
  for (int i = 0; i < 8; ++i) z.f[i] = 0.0f;
  return z.v;
}

__device__ __forceinline__ v8f wmma_bf16(v16bf a, v16bf b, v8f c) {
  return __builtin_amdgcn_wmma_f32_16x16x32_bf16(false, a, false, b, (short)0, c, false, false);
}

// A-fragment (16x32 bf16): caller passes base = &A[row_of_this_lane][k0].
// ISA layout: lane m=lane%16, half h=lane/16; vgpr i holds k = (i/4)*16 + h*8 + (i%4)*2 (+e)
__device__ __forceinline__ v16bf load_frag_a_row(const u16* rowbase) {
  const int h = (threadIdx.x & 31) >> 4;
  union { v16bf v; unsigned u[8]; } f;
#pragma unroll
  for (int i = 0; i < 8; ++i) {
    const int kk = ((i >> 2) << 4) + (h << 3) + ((i & 3) << 1);
    f.u[i] = *(const unsigned*)(rowbase + kk);
  }
  return f.v;
}

// B-fragment (32x16 bf16): caller passes base = &Bt[col_of_this_lane][k0] (Bt = B^T, K contiguous).
// Layout: lane n=lane%16, half h; vgpr i holds k = (i/4)*16 + h*8 + (i%4)*2 (+e)
__device__ __forceinline__ v16bf load_frag_b_col(const u16* colbase) {
  const int h = (threadIdx.x & 31) >> 4;
  union { v16bf v; unsigned u[8]; } f;
#pragma unroll
  for (int i = 0; i < 8; ++i) {
    const int kk = ((i >> 2) << 4) + (h << 3) + ((i & 3) << 1);
    f.u[i] = *(const unsigned*)(colbase + kk);
  }
  return f.v;
}

// ---------------- conversion / layout kernels ----------------
__global__ void to_bf16_kernel(const float* __restrict__ in, u16* __restrict__ out, size_t n) {
  size_t i = (size_t)blockIdx.x * blockDim.x + threadIdx.x;
  if (i < n) out[i] = f2bf(in[i]);
}

__global__ void shift_prev_kernel(const float* __restrict__ x, u16* __restrict__ out) {
  size_t i = (size_t)blockIdx.x * blockDim.x + threadIdx.x;
  if (i >= NXD) return;
  size_t bt = i / CD;
  size_t t = bt % CT;
  out[i] = (t == 0) ? (u16)0 : f2bf(x[i - CD]);
}

// W[K][N] f32 -> Wt[N][K] bf16
__global__ void transpose_bf16_kernel(const float* __restrict__ W, u16* __restrict__ Wt, int K, int N) {
  size_t i = (size_t)blockIdx.x * blockDim.x + threadIdx.x;
  if (i >= (size_t)K * N) return;
  size_t n = i / K, k = i % K;
  Wt[i] = f2bf(W[k * (size_t)N + n]);
}

// qkv_bf (B*T,3D) -> q/k/v [(b*H+h)*T + t][dh] bf16
__global__ void pack_qkv_kernel(const u16* __restrict__ qkv,
                                u16* __restrict__ q, u16* __restrict__ k, u16* __restrict__ v) {
  size_t i = (size_t)blockIdx.x * blockDim.x + threadIdx.x;
  if (i >= NBT * 3 * CD) return;
  int col = (int)(i % (3 * CD));
  size_t rbt = i / (3 * CD);
  int s = col / CD, rem = col % CD;
  int hh = rem / CDH, dh = rem % CDH;
  size_t b = rbt / CT, t = rbt % CT;
  size_t dst = (((b * CH + hh) * CT) + t) * CDH + dh;
  u16 val = qkv[i];
  if (s == 0) q[dst] = val; else if (s == 1) k[dst] = val; else v[dst] = val;
}

// ---------------- main WMMA GEMM: C[M][N] = A[M][K] @ Bt[N][K]^T (+bias) ----------------
template <bool OUT_BF16, bool HAS_BIAS>
__global__ void __launch_bounds__(256) gemm128_bf16(
    const u16* __restrict__ A, const u16* __restrict__ Bt,
    const float* __restrict__ bias, void* __restrict__ Cout, int M, int N, int K) {
  const int lane = threadIdx.x & 31;
  const int wv = threadIdx.x >> 5;       // 0..7
  const int wy = wv >> 2, wx = wv & 3;   // 2x4 wave grid
  const int ln = lane & 15, hf = lane >> 4;
  const int bm = blockIdx.y * 128, bn = blockIdx.x * 128;

  v8f acc[4][2];
#pragma unroll
  for (int i = 0; i < 4; ++i)
#pragma unroll
    for (int j = 0; j < 2; ++j) acc[i][j] = v8f_zero();

  const u16* Abase = A + (size_t)(bm + wy * 64 + ln) * K;
  const u16* Bbase = Bt + (size_t)(bn + wx * 32 + ln) * K;

  for (int k0 = 0; k0 < K; k0 += 32) {
    if (k0 + 32 < K) {
      __builtin_prefetch(Abase + k0 + 32, 0, 0);
      __builtin_prefetch(Bbase + k0 + 32, 0, 0);
    }
    v16bf af[4], bfg[2];
#pragma unroll
    for (int i = 0; i < 4; ++i) af[i] = load_frag_a_row(Abase + (size_t)i * 16 * K + k0);
#pragma unroll
    for (int j = 0; j < 2; ++j) bfg[j] = load_frag_b_col(Bbase + (size_t)j * 16 * K + k0);
#pragma unroll
    for (int i = 0; i < 4; ++i)
#pragma unroll
      for (int j = 0; j < 2; ++j) acc[i][j] = wmma_bf16(af[i], bfg[j], acc[i][j]);
  }

#pragma unroll
  for (int j = 0; j < 2; ++j) {
    const int n = bn + wx * 32 + j * 16 + ln;
    const float bv = HAS_BIAS ? bias[n] : 0.0f;
#pragma unroll
    for (int i = 0; i < 4; ++i) {
      union { v8f v; float f[8]; } c; c.v = acc[i][j];
#pragma unroll
      for (int r = 0; r < 8; ++r) {
        const int m = bm + wy * 64 + i * 16 + hf * 8 + r;
        const float val = c.f[r] + bv;
        if (OUT_BF16) ((u16*)Cout)[(size_t)m * N + n] = f2bf(val);
        else          ((float*)Cout)[(size_t)m * N + n] = val;
      }
    }
  }
}

// ---------------- narrow GEMM: N = NT*16 (<=64), one 16-row tile per wave ----------------
template <int NT, bool HAS_BIAS>
__global__ void __launch_bounds__(128) gemm_narrow_bf16(
    const u16* __restrict__ A, const u16* __restrict__ Bt,
    const float* __restrict__ bias, float* __restrict__ C, int K) {
  const int lane = threadIdx.x & 31;
  const int wv = threadIdx.x >> 5;
  const int ln = lane & 15, hf = lane >> 4;
  const int rowA = blockIdx.x * 64 + wv * 16 + ln;
  v8f acc[NT];
#pragma unroll
  for (int j = 0; j < NT; ++j) acc[j] = v8f_zero();
  const u16* Ab = A + (size_t)rowA * K;
  for (int k0 = 0; k0 < K; k0 += 32) {
    v16bf af = load_frag_a_row(Ab + k0);
#pragma unroll
    for (int j = 0; j < NT; ++j) {
      v16bf bf_ = load_frag_b_col(Bt + (size_t)(j * 16 + ln) * K + k0);
      acc[j] = wmma_bf16(af, bf_, acc[j]);
    }
  }
  const int N = NT * 16;
#pragma unroll
  for (int j = 0; j < NT; ++j) {
    const int n = j * 16 + ln;
    const float bv = HAS_BIAS ? bias[n] : 0.0f;
    union { v8f v; float f[8]; } c; c.v = acc[j];
#pragma unroll
    for (int r = 0; r < 8; ++r) {
      const int m = blockIdx.x * 64 + wv * 16 + hf * 8 + r;
      C[(size_t)m * N + n] = c.f[r] + bv;
    }
  }
}

// ---------------- flash attention (causal), WMMA bf16 ----------------
__global__ void __launch_bounds__(128) attn_kernel(
    const u16* __restrict__ q, const u16* __restrict__ k, const u16* __restrict__ v,
    float* __restrict__ seq_out) {
  const int bh = blockIdx.y;
  const int b = bh >> 4, hh = bh & 15;
  const int tile = blockIdx.x;          // 64-row q tile
  const int lane = threadIdx.x & 31;
  const int wv = threadIdx.x >> 5;      // 0..3, 16 q-rows each
  const int ln = lane & 15, hf = lane >> 4;

  __shared__ u16 Vt[64][34];            // transposed V chunk: [dh][key]
  __shared__ u16 Pl[4][16][34];         // per-wave P tile (16 rows x 32 keys)

  const u16* qb = q + (size_t)bh * CT * CDH;
  const u16* kb = k + (size_t)bh * CT * CDH;
  const u16* vb = v + (size_t)bh * CT * CDH;

  const int rowA = tile * 64 + wv * 16 + ln;
  const v16bf qf0 = load_frag_a_row(qb + (size_t)rowA * CDH);
  const v16bf qf1 = load_frag_a_row(qb + (size_t)rowA * CDH + 32);

  v8f O[4];
#pragma unroll
  for (int dt = 0; dt < 4; ++dt) O[dt] = v8f_zero();
  float mrow[8], lrow[8];
#pragma unroll
  for (int r = 0; r < 8; ++r) { mrow[r] = -1e30f; lrow[r] = 0.0f; }

  const int nch = tile * 2 + 2;         // causal: key chunks of 32
  for (int c = 0; c < nch; ++c) {
    const int ks = c * 32;
    __syncthreads();
    for (int e = threadIdx.x; e < 2048; e += 128) {   // stage V^T
      const int key = e >> 6, dh = e & 63;
      Vt[dh][key] = vb[(size_t)(ks + key) * CDH + dh];
    }
    __syncthreads();

    // S = q @ k^T for two 16-key subtiles
    union { v8f v; float f[8]; } s0, s1;
    {
      const u16* c0 = kb + (size_t)(ks + ln) * CDH;
      const u16* c1 = kb + (size_t)(ks + 16 + ln) * CDH;
      v8f t0 = wmma_bf16(qf0, load_frag_b_col(c0), v8f_zero());
      t0 = wmma_bf16(qf1, load_frag_b_col(c0 + 32), t0);
      v8f t1 = wmma_bf16(qf0, load_frag_b_col(c1), v8f_zero());
      t1 = wmma_bf16(qf1, load_frag_b_col(c1 + 32), t1);
      s0.v = t0; s1.v = t1;
    }

    // online softmax, P -> LDS (bf16)
#pragma unroll
    for (int r = 0; r < 8; ++r) {
      const int rowg = tile * 64 + wv * 16 + hf * 8 + r;
      const int c0g = ks + ln, c1g = ks + 16 + ln;
      float a0 = (c0g <= rowg) ? s0.f[r] * 0.125f : -1e30f;
      float a1 = (c1g <= rowg) ? s1.f[r] * 0.125f : -1e30f;
      float tmax = fmaxf(a0, a1);
      tmax = fmaxf(tmax, __shfl_xor(tmax, 8));
      tmax = fmaxf(tmax, __shfl_xor(tmax, 4));
      tmax = fmaxf(tmax, __shfl_xor(tmax, 2));
      tmax = fmaxf(tmax, __shfl_xor(tmax, 1));
      const float nm = fmaxf(mrow[r], tmax);
      const float corr = __expf(mrow[r] - nm);
      const float p0 = __expf(a0 - nm), p1 = __expf(a1 - nm);
      float ps = p0 + p1;
      ps += __shfl_xor(ps, 8); ps += __shfl_xor(ps, 4);
      ps += __shfl_xor(ps, 2); ps += __shfl_xor(ps, 1);
      lrow[r] = lrow[r] * corr + ps;
      mrow[r] = nm;
#pragma unroll
      for (int dt = 0; dt < 4; ++dt) O[dt][r] *= corr;
      Pl[wv][hf * 8 + r][ln] = f2bf(p0);
      Pl[wv][hf * 8 + r][16 + ln] = f2bf(p1);
    }

    // O += P @ V  (wave-private LDS, DS ops in-order within wave)
    const v16bf pf = load_frag_a_row(&Pl[wv][ln][0]);
#pragma unroll
    for (int dt = 0; dt < 4; ++dt) {
      const v16bf vf = load_frag_b_col(&Vt[dt * 16 + ln][0]);
      O[dt] = wmma_bf16(pf, vf, O[dt]);
    }
  }

#pragma unroll
  for (int dt = 0; dt < 4; ++dt) {
    union { v8f v; float f[8]; } o; o.v = O[dt];
#pragma unroll
    for (int r = 0; r < 8; ++r) {
      const int m = tile * 64 + wv * 16 + hf * 8 + r;
      const int dh = dt * 16 + ln;
      seq_out[((size_t)b * CT + m) * CD + hh * CDH + dh] = o.f[r] / lrow[r];
    }
  }
}

// ---------------- Plucker exterior products ----------------
__global__ void exterior_kernel(const float* __restrict__ p1, const float* __restrict__ p2,
                                float* __restrict__ out, int n, int asJw) {
  int i = blockIdx.x * blockDim.x + threadIdx.x;
  if (i >= n) return;
  const float* a = p1 + (size_t)i * 4;
  const float* b = p2 + (size_t)i * 4;
  const float L0 = a[0]*b[1] - a[1]*b[0];
  const float L1 = a[0]*b[2] - a[2]*b[0];
  const float L2 = a[0]*b[3] - a[3]*b[0];
  const float L3 = a[1]*b[2] - a[2]*b[1];
  const float L4 = a[1]*b[3] - a[3]*b[1];
  const float L5 = a[2]*b[3] - a[3]*b[2];
  const float nrm = sqrtf(L0*L0 + L1*L1 + L2*L2 + L3*L3 + L4*L4 + L5*L5);
  const float inv = 1.0f / fmaxf(nrm, 1e-12f);
  float* o = out + (size_t)i * 6;
  if (asJw) { // fused: lines @ J6 (signed permutation)
    o[0] =  L5*inv; o[1] = -L4*inv; o[2] =  L3*inv;
    o[3] =  L2*inv; o[4] = -L1*inv; o[5] =  L0*inv;
  } else {
    o[0] = L0*inv; o[1] = L1*inv; o[2] = L2*inv;
    o[3] = L3*inv; o[4] = L4*inv; o[5] = L5*inv;
  }
}

// ---------------- Riccati scan: 8 lanes per (b,h), one M-column per lane ----------------
__global__ void __launch_bounds__(256) scan_kernel(
    const float* __restrict__ rd, const float* __restrict__ jw,
    const float* __restrict__ Amat, float* __restrict__ score) {
  const int lane = threadIdx.x & 31;
  const int wv = threadIdx.x >> 5;            // 0..7
  const int lig = lane & 7;                   // lane-in-group
  const int grp = (wv << 2) | (lane >> 3);    // 0..31 == b*H + h
  const int b = grp >> 4, hh = grp & 15;
  const int col = (lig < 6) ? lig : 0;
  const float act = (lig < 6) ? 1.0f : 0.0f;

  float Af[36], Mc[6];
#pragma unroll
  for (int i = 0; i < 36; ++i) Af[i] = Amat[hh * 36 + i];
#pragma unroll
  for (int i = 0; i < 6; ++i) Mc[i] = 0.0f;

  for (int t = 0; t < CT; ++t) {
    const float* rp = rd + (((size_t)b * CT + t) * CH + hh) * 6;
    const float* jp = jw + (((size_t)b * CT + t) * CH + hh) * 6;
    float rv[6], jv[6];
#pragma unroll
    for (int i = 0; i < 6; ++i) { rv[i] = rp[i]; jv[i] = jp[i]; }
    // score = r^T M r, with M from previous step
    float cs = 0.0f;
#pragma unroll
    for (int i = 0; i < 6; ++i) cs += rv[i] * Mc[i];     // sum_p rv[p] * M[p][col]
    float contrib = act * rv[col] * cs;
    contrib += __shfl_xor(contrib, 4);
    contrib += __shfl_xor(contrib, 2);
    contrib += __shfl_xor(contrib, 1);
    if (lig == 0) score[((size_t)b * CT + t) * CH + hh] = contrib;
    // AM[:,col] = A * M[:,col]
    float AMc[6];
#pragma unroll
    for (int i = 0; i < 6; ++i) {
      float s = 0.0f;
#pragma unroll
      for (int r2 = 0; r2 < 6; ++r2) s += Af[i * 6 + r2] * Mc[r2];
      AMc[i] = s;
    }
    // M'[:,col] = sum_r AM[:,r] * A[col][r] + jw * jw[col]
    float Mn[6];
#pragma unroll
    for (int i = 0; i < 6; ++i) Mn[i] = jv[i] * jv[col];
#pragma unroll
    for (int r2 = 0; r2 < 6; ++r2) {
      const float ajr = Af[col * 6 + r2];
#pragma unroll
      for (int i = 0; i < 6; ++i) Mn[i] += __shfl(AMc[i], r2, 8) * ajr;
    }
#pragma unroll
    for (int i = 0; i < 6; ++i) Mc[i] = Mn[i];
  }
}

// ---------------- gating + combine ----------------
__device__ __forceinline__ float sigmoidf(float x) { return 1.0f / (1.0f + __expf(-x)); }

__global__ void gate_kernel(const float* __restrict__ score, const float* __restrict__ glin,
                            const float* __restrict__ mem_scale, float* __restrict__ gated) {
  int i = blockIdx.x * blockDim.x + threadIdx.x;
  if (i >= (int)NBT) return;
  float acc = 0.0f;
#pragma unroll
  for (int h = 0; h < CH; ++h)
    acc += sigmoidf(score[(size_t)i * CH + h] * mem_scale[h]) * sigmoidf(glin[(size_t)i * CH + h]);
  gated[i] = acc * (1.0f / CH);
}

__global__ void combine_kernel(const float* __restrict__ seq_out, const float* __restrict__ mem_val,
                               const float* __restrict__ gated, u16* __restrict__ combined) {
  size_t i = (size_t)blockIdx.x * blockDim.x + threadIdx.x;
  if (i >= NXD) return;
  const size_t row = i / CD;
  combined[i] = f2bf(seq_out[i] + gated[row] * mem_val[i]);
}

// ---------------- host orchestration ----------------
extern "C" void kernel_launch(void* const* d_in, const int* in_sizes, int n_in,
                              void* d_out, int out_size, void* d_ws, size_t ws_size,
                              hipStream_t stream) {
  (void)in_sizes; (void)n_in; (void)out_size; (void)ws_size;
  const float* x    = (const float*)d_in[0];
  const float* Wqkv = (const float*)d_in[1];
  const float* bqkv = (const float*)d_in[2];
  const float* W1w  = (const float*)d_in[3];
  const float* W2w  = (const float*)d_in[4];
  const float* W1r  = (const float*)d_in[5];
  const float* W2r  = (const float*)d_in[6];
  const float* Wmv  = (const float*)d_in[7];
  const float* bmv  = (const float*)d_in[8];
  const float* Wg   = (const float*)d_in[9];
  const float* bg   = (const float*)d_in[10];
  const float* mem_scale = (const float*)d_in[11];
  const float* Wout = (const float*)d_in[12];
  const float* bout = (const float*)d_in[13];
  const float* Amat = (const float*)d_in[14];
  float* out = (float*)d_out;

  size_t off = 0;
  auto alloc = [&](size_t bytes) -> void* {
    void* p = (char*)d_ws + off;
    off += (bytes + 255) & ~(size_t)255;
    return p;
  };

  u16* xb     = (u16*)alloc(NXD * 2);
  u16* xpb    = (u16*)alloc(NXD * 2);
  u16* Wqkv_t = (u16*)alloc((size_t)3 * CD * CD * 2);
  u16* Wmv_t  = (u16*)alloc((size_t)CD * CD * 2);
  u16* Wout_t = (u16*)alloc((size_t)CD * CD * 2);
  u16* W1w_t  = (u16*)alloc((size_t)64 * CD * 2);
  u16* W2w_t  = (u16*)alloc((size_t)64 * CD * 2);
  u16* W1r_t  = (u16*)alloc((size_t)64 * CD * 2);
  u16* W2r_t  = (u16*)alloc((size_t)64 * CD * 2);
  u16* Wg_t   = (u16*)alloc((size_t)16 * CD * 2);
  char* qkvreg = (char*)alloc(NXD * 3 * 2);   // 24 MB; reused after pack
  u16*   qkv_bf   = (u16*)qkvreg;
  float* mem_val  = (float*)qkvreg;                        // 16 MB overlay
  u16*   combined = (u16*)(qkvreg + NXD * 4);              // 8 MB overlay
  u16* qpk = (u16*)alloc(NXD * 2);
  u16* kpk = (u16*)alloc(NXD * 2);
  u16* vpk = (u16*)alloc(NXD * 2);
  float* seq_out = (float*)alloc(NXD * 4);
  float* w1 = (float*)alloc(NBT * 64 * 4);
  float* w2 = (float*)alloc(NBT * 64 * 4);
  float* r1 = (float*)alloc(NBT * 64 * 4);
  float* r2 = (float*)alloc(NBT * 64 * 4);
  float* rdl = (float*)alloc(NBT * CH * 6 * 4);
  float* jwl = (float*)alloc(NBT * CH * 6 * 4);
  float* glin   = (float*)alloc(NBT * CH * 4);
  float* mscore = (float*)alloc(NBT * CH * 4);
  float* gated  = (float*)alloc(NBT * 4);

  const int TPB = 256;
  auto blk = [](size_t n, int t) { return (unsigned)((n + t - 1) / t); };

  // 1) precision / layout prep
  to_bf16_kernel<<<blk(NXD, TPB), TPB, 0, stream>>>(x, xb, NXD);
  shift_prev_kernel<<<blk(NXD, TPB), TPB, 0, stream>>>(x, xpb);
  transpose_bf16_kernel<<<blk((size_t)CD * 3 * CD, TPB), TPB, 0, stream>>>(Wqkv, Wqkv_t, CD, 3 * CD);
  transpose_bf16_kernel<<<blk((size_t)CD * CD, TPB), TPB, 0, stream>>>(Wmv, Wmv_t, CD, CD);
  transpose_bf16_kernel<<<blk((size_t)CD * CD, TPB), TPB, 0, stream>>>(Wout, Wout_t, CD, CD);
  transpose_bf16_kernel<<<blk((size_t)CD * 64, TPB), TPB, 0, stream>>>(W1w, W1w_t, CD, 64);
  transpose_bf16_kernel<<<blk((size_t)CD * 64, TPB), TPB, 0, stream>>>(W2w, W2w_t, CD, 64);
  transpose_bf16_kernel<<<blk((size_t)CD * 64, TPB), TPB, 0, stream>>>(W1r, W1r_t, CD, 64);
  transpose_bf16_kernel<<<blk((size_t)CD * 64, TPB), TPB, 0, stream>>>(W2r, W2r_t, CD, 64);
  transpose_bf16_kernel<<<blk((size_t)CD * 16, TPB), TPB, 0, stream>>>(Wg, Wg_t, CD, 16);

  // 2) qkv GEMM (bf16 out) + head pack
  gemm128_bf16<true, true><<<dim3(3 * CD / 128, NBT / 128), 256, 0, stream>>>(
      xb, Wqkv_t, bqkv, qkv_bf, (int)NBT, 3 * CD, CD);
  pack_qkv_kernel<<<blk(NBT * 3 * CD, TPB), TPB, 0, stream>>>(qkv_bf, qpk, kpk, vpk);

  // 3) mem_val GEMM (overwrites qkv region AFTER pack)
  gemm128_bf16<false, true><<<dim3(CD / 128, NBT / 128), 256, 0, stream>>>(
      xb, Wmv_t, bmv, mem_val, (int)NBT, CD, CD);

  // 4) causal flash attention
  attn_kernel<<<dim3(CT / 64, CB * CH), 128, 0, stream>>>(qpk, kpk, vpk, seq_out);

  // 5) line projections + exterior products
  gemm_narrow_bf16<4, false><<<NBT / 64, 128, 0, stream>>>(xpb, W1w_t, nullptr, w1, CD);
  gemm_narrow_bf16<4, false><<<NBT / 64, 128, 0, stream>>>(xb,  W2w_t, nullptr, w2, CD);
  gemm_narrow_bf16<4, false><<<NBT / 64, 128, 0, stream>>>(xb,  W1r_t, nullptr, r1, CD);
  gemm_narrow_bf16<4, false><<<NBT / 64, 128, 0, stream>>>(xb,  W2r_t, nullptr, r2, CD);
  gemm_narrow_bf16<1, true ><<<NBT / 64, 128, 0, stream>>>(xb,  Wg_t,  bg, glin, CD);
  const int nlines = (int)(NBT * CH);
  exterior_kernel<<<blk(nlines, TPB), TPB, 0, stream>>>(w1, w2, jwl, nlines, 1); // write lines -> Jw
  exterior_kernel<<<blk(nlines, TPB), TPB, 0, stream>>>(r1, r2, rdl, nlines, 0); // read lines

  // 6) sequential Riccati scan (32 groups x 8 lanes, one block)
  scan_kernel<<<1, 256, 0, stream>>>(rdl, jwl, Amat, mscore);

  // 7) gated readout + output projection
  gate_kernel<<<blk(NBT, TPB), TPB, 0, stream>>>(mscore, glin, mem_scale, gated);
  combine_kernel<<<blk(NXD, TPB), TPB, 0, stream>>>(seq_out, mem_val, gated, combined);
  gemm128_bf16<false, true><<<dim3(CD / 128, NBT / 128), 256, 0, stream>>>(
      combined, Wout_t, bout, out, (int)NBT, CD, CD);
}
